// TransformerBlock_29368986370126
// MI455X (gfx1250) — matrix operand order repaired and checked by reading, not compiled, not run on hardware
//
#include <hip/hip_runtime.h>
#include <hip/hip_bf16.h>

typedef __attribute__((ext_vector_type(16))) _Float16 v16h;
typedef __attribute__((ext_vector_type(8)))  _Float16 v8h;
typedef __attribute__((ext_vector_type(8)))  float    v8f;

#define NB 2048
#define TT 128
#define CC 128
#define NH 4
#define HS 32
#define FFD 512

#define LDH 136   // f16 row stride (halves): 272 B -> conflict-free fragment loads
#define LDF 132   // f32 row stride (floats): 528 B

// ---- dynamic LDS layout (bytes), total 309248 B < 320 KB/WGP ----
#define OFF_XS   0        // float[128*132]  x residual / x1 (persistent)
#define OFF_HB   67584    // f16 [128*136]   h(LN1) -> attn out -> h2(LN2)   (A)
#define OFF_QB   102400   // f16             Q (A) | WprojT | W1T chunk      (B)
#define OFF_KB   137216   // f16             K [s][d] (B^T) | W2T chunk      (B)
#define OFF_VT   172032   // f16             V^T [d][s] (B)| ReLU FF chunk(A)
#define OFF_SB   206848   // float[128*132]  scores | f16 weight staging (QKV)
#define OFF_AB   274432   // f16             softmax probs (A)
#define SMEM_BYTES 309248

// One 16x16 f32 tile of A(16xK) x B(Kx16) with B stored TRANSPOSED [n][k].
// A row-major [m][k]. Both f16 in LDS. Fragment layouts per CDNA5 ISA §7.12.2.
// Per WMMA: 2x ds_load_b128 (A) + 2x ds_load_b128 (B), no packing VALU.
__device__ __forceinline__ v8f mma_tileT(const _Float16* __restrict__ A, int lda,
                                         const _Float16* __restrict__ Bt, int ldb,
                                         int K, v8f acc, int lane) {
  const int hf = lane >> 4;   // lane half
  const int l  = lane & 15;
  for (int k0 = 0; k0 < K; k0 += 32) {
    union { v16h v; v8h p[2]; } af, bf;
    af.p[0] = *(const v8h*)(A + l * lda + k0 + 8 * hf);
    af.p[1] = *(const v8h*)(A + l * lda + k0 + 16 + 8 * hf);
    bf.p[0] = *(const v8h*)(Bt + l * ldb + k0 + 8 * hf);
    bf.p[1] = *(const v8h*)(Bt + l * ldb + k0 + 16 + 8 * hf);
    acc = __builtin_amdgcn_wmma_f32_16x16x32_f16(false, af.v, false, bf.v,
                                                 (short)0, acc, false, false);
  }
  return acc;
}

// Row-major f16 store of a C/D tile (elem r of lane -> row r+8*hf, col l).
__device__ __forceinline__ void store_tile_rm(_Float16* D, int ldd, v8f acc, int lane) {
  union { v8f v; float f[8]; } a; a.v = acc;
  const int l = lane & 15, hf = lane >> 4;
#pragma unroll
  for (int r = 0; r < 8; ++r)
    D[(r + 8 * hf) * ldd + l] = (_Float16)a.f[r];
}

// Transposed f16 store: D indexed [n][m]; lane's 8 rows are contiguous -> 1x b128.
__device__ __forceinline__ void store_tile_tr(_Float16* D, int ldd, v8f acc, int lane) {
  union { v8f v; float f[8]; } a; a.v = acc;
  union { v8h v; _Float16 h[8]; } st;
#pragma unroll
  for (int r = 0; r < 8; ++r) st.h[r] = (_Float16)a.f[r];
  const int l = lane & 15, hf = lane >> 4;
  *(v8h*)(D + l * ldd + 8 * hf) = st.v;
}

__device__ __forceinline__ void layer_norm_rows(const float* __restrict__ xs,
                                                const float* __restrict__ w,
                                                const float* __restrict__ bvec,
                                                _Float16* __restrict__ dst, int tid) {
  if (tid < TT) {
    const float* row = xs + tid * LDF;
    float m = 0.f;
#pragma unroll 4
    for (int c = 0; c < CC; ++c) m += row[c];
    m *= (1.0f / CC);
    float var = 0.f;
#pragma unroll 4
    for (int c = 0; c < CC; ++c) { float d = row[c] - m; var += d * d; }
    var *= (1.0f / CC);
    float rs = rsqrtf(var + 1e-5f);
#pragma unroll 4
    for (int c = 0; c < CC; ++c)
      dst[tid * LDH + c] = (_Float16)((row[c] - m) * rs * w[c] + bvec[c]);
  }
}

__global__ __launch_bounds__(256) void txblock_kernel(
    const float* __restrict__ x,
    const float* __restrict__ ln1_w, const float* __restrict__ ln1_b,
    const float* __restrict__ ln2_w, const float* __restrict__ ln2_b,
    const float* __restrict__ Wq, const float* __restrict__ Wk,
    const float* __restrict__ Wv, const float* __restrict__ Wproj,
    const float* __restrict__ bproj,
    const float* __restrict__ W1, const float* __restrict__ b1,
    const float* __restrict__ W2, const float* __restrict__ b2,
    float* __restrict__ out)
{
  extern __shared__ __align__(16) char smem[];
  float*    xs = (float*)(smem + OFF_XS);
  _Float16* hb = (_Float16*)(smem + OFF_HB);
  _Float16* qb = (_Float16*)(smem + OFF_QB);
  _Float16* kb = (_Float16*)(smem + OFF_KB);
  _Float16* vt = (_Float16*)(smem + OFF_VT);
  float*    sb = (float*)(smem + OFF_SB);
  _Float16* wl = (_Float16*)(smem + OFF_SB);   // aliases sb (disjoint phases)
  _Float16* ab = (_Float16*)(smem + OFF_AB);

  const int bb   = blockIdx.x;
  const int tid  = threadIdx.x;
  const int lane = tid & 31;
  const int wv   = tid >> 5;       // wave id 0..7 -> owns output cols wv*16..+15
  const int l    = lane & 15;
  const int hf   = lane >> 4;

  // ---- warm L2 for late-phase weights (global_prefetch_b8, fire-and-forget) ----
  {
    // W1 + W2: 65536 f32 each -> 2048 x 128B lines each; Wproj: 512 lines.
    for (int i = tid; i < 2048; i += 256) {
      __builtin_prefetch(W1 + i * 32, 0, 1);
      __builtin_prefetch(W2 + i * 32, 0, 1);
    }
    for (int i = tid; i < 512; i += 256)
      __builtin_prefetch(Wproj + i * 32, 0, 1);
  }

  // ---- stream x[bb] (128x128 f32) into padded LDS rows ----
  {
    const float4* xg = (const float4*)(x + (size_t)bb * (TT * CC));
    for (int i = tid; i < TT * (CC / 4); i += 256) {
      int r = i >> 5, c4 = i & 31;                 // 32 float4 per row
      *((float4*)(xs + r * LDF) + c4) = xg[r * 32 + c4];
    }
  }
  __syncthreads();

  // ---- LN1 ----
  layer_norm_rows(xs, ln1_w, ln1_b, hb, tid);
  __syncthreads();

  // ---- QKV: stage W^T [n][c] f16, then Q/K row-major, V transposed ----
#pragma unroll
  for (int m = 0; m < 3; ++m) {
    const float* Wg = (m == 0) ? Wq : (m == 1) ? Wk : Wv;
    for (int i = tid; i < CC * CC; i += 256) {
      int n = i >> 7, c = i & 127;                 // wl[n][c] = W[h(n)][c][d(n)]
      wl[n * LDH + c] = (_Float16)Wg[((n >> 5) << 12) + (c << 5) + (n & 31)];
    }
    __syncthreads();
#pragma unroll
    for (int i = 0; i < 8; ++i) {
      v8f acc = {};
      acc = mma_tileT(hb + i * 16 * LDH, LDH, wl + (wv * 16) * LDH, LDH, CC, acc, lane);
      if (m == 0)      store_tile_rm(qb + (i * 16) * LDH + wv * 16, LDH, acc, lane);
      else if (m == 1) store_tile_rm(kb + (i * 16) * LDH + wv * 16, LDH, acc, lane);
      else             store_tile_tr(vt + (wv * 16) * LDH + i * 16, LDH, acc, lane);
    }
    __syncthreads();
  }

  // ---- attention, head by head ----
  const float scale = 0.088388347648318447f;   // C^-0.5 (ref scales by full C)
  for (int h = 0; h < NH; ++h) {
    // scores = Q_h (128x32) @ K_h^T : K stored [s][d] == B^T layout
#pragma unroll
    for (int i = 0; i < 8; ++i) {
      v8f acc = {};
      acc = mma_tileT(qb + i * 16 * LDH + h * HS, LDH,
                      kb + (wv * 16) * LDH + h * HS, LDH, HS, acc, lane);
      union { v8f v; float f[8]; } a; a.v = acc;
#pragma unroll
      for (int r = 0; r < 8; ++r) {
        int t = i * 16 + r + 8 * hf, s = wv * 16 + l;
        sb[t * LDF + s] = (s <= t) ? a.f[r] * scale : -INFINITY;
      }
    }
    __syncthreads();
    // row softmax -> f16 probs (row-major: A operand of P@V)
    if (tid < TT) {
      const float* row = sb + tid * LDF;
      float mx = -INFINITY;
      for (int s = 0; s < TT; ++s) mx = fmaxf(mx, row[s]);
      float sum = 0.f;
      for (int s = 0; s < TT; ++s) sum += __expf(row[s] - mx);
      float inv = 1.0f / sum;
      for (int s = 0; s < TT; ++s)
        ab[tid * LDH + s] = (_Float16)(__expf(row[s] - mx) * inv);
    }
    __syncthreads();
    // O_h = P (128x128) @ V_h : V stored transposed [d][s] == B^T layout
#pragma unroll
    for (int u = 0; u < 2; ++u) {
      v8f acc = {};
      acc = mma_tileT(ab + (wv * 16) * LDH, LDH,
                      vt + (h * HS + u * 16) * LDH, LDH, TT, acc, lane);
      store_tile_rm(hb + (wv * 16) * LDH + h * HS + u * 16, LDH, acc, lane);
    }
    __syncthreads();
  }

  // ---- output projection + residual into xs (WprojT staged in qb region) ----
  for (int i = tid; i < CC * CC; i += 256) {
    int n = i >> 7, c = i & 127;
    qb[n * LDH + c] = (_Float16)Wproj[c * CC + n];
  }
  __syncthreads();
#pragma unroll
  for (int i = 0; i < 8; ++i) {
    v8f acc = {};
    acc = mma_tileT(hb + i * 16 * LDH, LDH, qb + (wv * 16) * LDH, LDH, CC, acc, lane);
    union { v8f v; float f[8]; } a; a.v = acc;
    int col = wv * 16 + l;
    float bp = bproj[col];
#pragma unroll
    for (int r = 0; r < 8; ++r)
      xs[(i * 16 + r + 8 * hf) * LDF + col] += a.f[r] + bp;
  }
  __syncthreads();

  // ---- LN2 ----
  layer_norm_rows(xs, ln2_w, ln2_b, hb, tid);
  __syncthreads();

  // ---- MLP: relu(h2 @ W1 + b1) @ W2, 4 K-chunks of 128, acc in registers ----
  v8f facc[8];
#pragma unroll
  for (int i = 0; i < 8; ++i) { v8f z = {}; facc[i] = z; }
  _Float16* w1t = qb;     // W1 chunk, transposed [n][c]
  _Float16* w2t = kb;     // W2 chunk, transposed [n][k]
  _Float16* ffc = vt;     // ReLU intermediate, row-major (A operand)
  for (int j = 0; j < 4; ++j) {
    for (int i = tid; i < CC * CC; i += 256) {
      int n = i >> 7, c = i & 127;
      w1t[n * LDH + c] = (_Float16)W1[c * FFD + j * 128 + n];
    }
    for (int i = tid; i < CC * CC; i += 256) {
      int n = i >> 7, k = i & 127;
      w2t[n * LDH + k] = (_Float16)W2[(j * 128 + k) * CC + n];
    }
    __syncthreads();
#pragma unroll
    for (int i = 0; i < 8; ++i) {
      v8f acc = {};
      acc = mma_tileT(hb + i * 16 * LDH, LDH, w1t + (wv * 16) * LDH, LDH, CC, acc, lane);
      union { v8f v; float f[8]; } a; a.v = acc;
      int col = wv * 16 + l;
      float bv = b1[j * 128 + col];
#pragma unroll
      for (int r = 0; r < 8; ++r)
        ffc[(i * 16 + r + 8 * hf) * LDH + col] = (_Float16)fmaxf(a.f[r] + bv, 0.f);
    }
    __syncthreads();
#pragma unroll
    for (int i = 0; i < 8; ++i)
      facc[i] = mma_tileT(ffc + i * 16 * LDH, LDH, w2t + (wv * 16) * LDH, LDH, CC,
                          facc[i], lane);
    __syncthreads();
  }

  // ---- epilogue: out = x1 + ff + b2 ----
  {
    int col = wv * 16 + l;
    float b2v = b2[col];
    float* op = out + (size_t)bb * (TT * CC);
#pragma unroll
    for (int i = 0; i < 8; ++i) {
      union { v8f v; float f[8]; } a; a.v = facc[i];
#pragma unroll
      for (int r = 0; r < 8; ++r) {
        int row = i * 16 + r + 8 * hf;
        op[row * CC + col] = xs[row * LDF + col] + a.f[r] + b2v;
      }
    }
  }
}

extern "C" void kernel_launch(void* const* d_in, const int* in_sizes, int n_in,
                              void* d_out, int out_size, void* d_ws, size_t ws_size,
                              hipStream_t stream) {
  (void)in_sizes; (void)n_in; (void)out_size; (void)d_ws; (void)ws_size;
  const float* x     = (const float*)d_in[0];
  const float* ln1_w = (const float*)d_in[1];
  const float* ln1_b = (const float*)d_in[2];
  const float* ln2_w = (const float*)d_in[3];
  const float* ln2_b = (const float*)d_in[4];
  const float* Wq    = (const float*)d_in[5];
  const float* Wk    = (const float*)d_in[6];
  const float* Wv    = (const float*)d_in[7];
  const float* Wpr   = (const float*)d_in[8];
  const float* bpr   = (const float*)d_in[9];
  const float* W1    = (const float*)d_in[10];
  const float* b1    = (const float*)d_in[11];
  const float* W2    = (const float*)d_in[12];
  const float* b2    = (const float*)d_in[13];
  float* out = (float*)d_out;
  txblock_kernel<<<dim3(NB), dim3(256), SMEM_BYTES, stream>>>(
      x, ln1_w, ln1_b, ln2_w, ln2_b, Wq, Wk, Wv, Wpr, bpr, W1, b1, W2, b2, out);
}
